// DeepseekV3MoE_11613591568697
// MI455X (gfx1250) — compile-verified, matrix operand order checked by reading, not run on hardware
//
#include <hip/hip_runtime.h>
#include <math.h>

// ---------------- problem constants ----------------
#define T_TOK   2048
#define H_DIM   1024
#define N_EXP   16
#define I_MOE   512
#define I_SH    1024      // N_SH * I_MOE
#define SCALE_F 2.5f

// workspace layout (bf16 element offsets)
#define OFF_GATE 0ull
#define OFF_UP   8388608ull        // 16*512*1024
#define OFF_DOWN 16777216ull
#define OFF_SHG  25165824ull       // + 1024*1024
#define OFF_SHU  26214400ull
#define OFF_SHD  27262976ull
#define OFF_X    28311552ull       // + 2048*1024
#define TOTAL_BF 30408704ull
#define COMB_OFF_BYTES (TOTAL_BF * 2ull)   // float combine weights after bf16 pool

// ---------------- CDNA5 WMMA types ----------------
typedef __attribute__((ext_vector_type(16))) __bf16 bf16x16;
typedef __attribute__((ext_vector_type(8)))  __bf16 bf16x8;
typedef __attribute__((ext_vector_type(8)))  float  f32x8;

static __device__ __forceinline__ unsigned short f2bf_bits(float f) {
    unsigned u = __builtin_bit_cast(unsigned, f);
    u += 0x7FFFu + ((u >> 16) & 1u);          // round to nearest even
    return (unsigned short)(u >> 16);
}
static __device__ __forceinline__ unsigned pack2bf(float lo, float hi) {
    return ((unsigned)f2bf_bits(hi) << 16) | f2bf_bits(lo);
}

static __device__ __forceinline__ f32x8 wmma_bf16(bf16x16 a, bf16x16 b, f32x8 c) {
    return __builtin_amdgcn_wmma_f32_16x16x32_bf16(
        false, a, false, b, (short)0, c, false, false);
}

// Fragment loader (global or LDS bf16): two b128 loads, no conversion VALU.
// p = base + row*stride + k0 + half*8 ; elems [0..8) at p, [8..16) at p+16.
static __device__ __forceinline__ bf16x16 ld_frag(const unsigned short* p) {
    bf16x8 lo = *(const bf16x8*)(p);
    bf16x8 hi = *(const bf16x8*)(p + 16);
    bf16x16 f;
#pragma unroll
    for (int i = 0; i < 8; ++i) { f[i] = lo[i]; f[8 + i] = hi[i]; }
    return f;
}

// =====================================================================
// Kernel 0: bulk f32 -> bf16 conversion (memory bound, once per launch)
// =====================================================================
__global__ void cvt_bf16_kernel(const float* __restrict__ src,
                                unsigned short* __restrict__ dst, int n8) {
    for (int i = blockIdx.x * blockDim.x + threadIdx.x; i < n8;
         i += gridDim.x * blockDim.x) {
        const float4* s = (const float4*)(src + (size_t)i * 8);
        float4 a = s[0], b = s[1];
        uint4 o;
        o.x = pack2bf(a.x, a.y); o.y = pack2bf(a.z, a.w);
        o.z = pack2bf(b.x, b.y); o.w = pack2bf(b.z, b.w);
        *(uint4*)(dst + (size_t)i * 8) = o;
    }
}

// =====================================================================
// Kernel 1: router — sigmoid scores, grouped top-k, combine weights [T,16]
// =====================================================================
__global__ void moe_router_kernel(const float* __restrict__ x,
                                  const float* __restrict__ rw,
                                  const float* __restrict__ ebias,
                                  float* __restrict__ comb) {
    const int lane = threadIdx.x & 31;
    const int wave = threadIdx.x >> 5;
    const int t    = blockIdx.x * 8 + wave;
    const int e    = lane & 15;

    const float* xr = x  + (size_t)t * H_DIM;
    const float* wr = rw + (size_t)e * H_DIM;
    float acc = 0.f;
#pragma unroll 4
    for (int h = 0; h < H_DIM; h += 4) {
        float4 xv = *(const float4*)(xr + h);
        float4 wv = *(const float4*)(wr + h);
        acc = fmaf(xv.x, wv.x, acc); acc = fmaf(xv.y, wv.y, acc);
        acc = fmaf(xv.z, wv.z, acc); acc = fmaf(xv.w, wv.w, acc);
    }
    float score = 1.f / (1.f + __expf(-acc));

    float s[16], scb[16];
#pragma unroll
    for (int i = 0; i < 16; ++i) {
        s[i]   = __shfl(score, i, 32);
        scb[i] = s[i] + ebias[i];
    }
    float gs[4];
#pragma unroll
    for (int g = 0; g < 4; ++g) {
        float m1 = -1e30f, m2 = -1e30f;
#pragma unroll
        for (int j = 0; j < 4; ++j) {
            float v = scb[4 * g + j];
            if (v > m1) { m2 = m1; m1 = v; } else if (v > m2) { m2 = v; }
        }
        gs[g] = m1 + m2;
    }
    int g1 = 0;
#pragma unroll
    for (int g = 1; g < 4; ++g) if (gs[g] > gs[g1]) g1 = g;
    int g2 = (g1 == 0) ? 1 : 0;
#pragma unroll
    for (int g = 0; g < 4; ++g) if (g != g1 && gs[g] > gs[g2]) g2 = g;

    float msk[16];
#pragma unroll
    for (int i = 0; i < 16; ++i) {
        int gi = i >> 2;
        msk[i] = (gi == g1 || gi == g2) ? scb[i] : 0.0f;
    }
    int idxs[4]; float tsum = 0.f;
#pragma unroll
    for (int k = 0; k < 4; ++k) {
        int bi = 0; float bv = -1e30f;
#pragma unroll
        for (int i = 0; i < 16; ++i) if (msk[i] > bv) { bv = msk[i]; bi = i; }
        idxs[k] = bi; msk[bi] = -1e30f; tsum += s[bi];
    }
    tsum += 1e-20f;
    float outw = 0.f;
#pragma unroll
    for (int k = 0; k < 4; ++k)
        if (idxs[k] == e) outw = s[e] / tsum * SCALE_F;
    if (lane < 16) comb[(size_t)t * N_EXP + e] = outw;
}

// =====================================================================
// Kernel 2: fused MoE FFN, all-bf16 data path.
// 1 block = 16 tokens, 8 waves; X tile cached in LDS across 16 experts
// + shared expert (2 K-halves); output accumulated in registers, one store.
// All fragment loads of a k-step are issued as one batch before the WMMA
// batch so later WMMAs overlap the memory latency (in-order VMEM returns
// let the compiler use partial s_wait_loadcnt).
// =====================================================================
#define XS (H_DIM + 8)   // LDS row strides padded: 4-bank step per row
#define HS (I_MOE + 8)

__global__ void __launch_bounds__(256)
moe_ffn_kernel(const unsigned short* __restrict__ xbf,
               const float* __restrict__ comb,
               const unsigned short* __restrict__ gate_w,
               const unsigned short* __restrict__ up_w,
               const unsigned short* __restrict__ down_w,
               const unsigned short* __restrict__ shg,
               const unsigned short* __restrict__ shu,
               const unsigned short* __restrict__ shd,
               float* __restrict__ out) {
    __shared__ unsigned short x_lds[16 * XS];   // 33.0 KB
    __shared__ unsigned short h_lds[16 * HS];   // 16.6 KB
    __shared__ float          comb_lds[16 * N_EXP];

    const int tid  = threadIdx.x;
    const int wave = tid >> 5;
    const int lane = tid & 31;
    const int r    = lane & 15;        // fragment row/col index
    const int half = lane >> 4;        // k-half select
    const int t0   = blockIdx.x * 16;

    // ---- stage X tile [16 x 1024] bf16 -> LDS (pure b128 copies) ----
    for (int i = tid * 8; i < 16 * H_DIM; i += 256 * 8) {
        int row = i >> 10, col = i & (H_DIM - 1);
        uint4 v = *(const uint4*)(xbf + (size_t)t0 * H_DIM + i);
        *(uint4*)&x_lds[row * XS + col] = v;
    }
    comb_lds[tid & 255] = comb[(size_t)t0 * N_EXP + (tid & 255)];
    __syncthreads();

    f32x8 oacc[8];
#pragma unroll
    for (int j = 0; j < 8; ++j) oacc[j] = (f32x8){};

    for (int ex = 0; ex < 18; ++ex) {          // 16 routed + 2 shared halves
        const unsigned short *GW, *UW, *DW;
        int drs, koff;
        if (ex < 16) {
            GW = gate_w + (size_t)ex * I_MOE * H_DIM;
            UW = up_w   + (size_t)ex * I_MOE * H_DIM;
            DW = down_w + (size_t)ex * H_DIM * I_MOE;
            drs = I_MOE; koff = 0;
        } else {
            int sh = ex - 16;
            GW = shg + (size_t)sh * I_MOE * H_DIM;
            UW = shu + (size_t)sh * I_MOE * H_DIM;
            DW = shd;
            drs = I_SH; koff = sh * I_MOE;
        }

        // ---------- Phase A: g,u for I-range [wave*64, +64) ----------
        const int i0 = wave * 64;
        f32x8 gacc[4], uacc[4];
#pragma unroll
        for (int j = 0; j < 4; ++j) { gacc[j] = (f32x8){}; uacc[j] = (f32x8){}; }

        for (int k0 = 0; k0 < H_DIM; k0 += 32) {
            // batch-issue all 16 weight-fragment loads for this k-step
            bf16x16 bfr[8];
#pragma unroll
            for (int j = 0; j < 4; ++j) {
                const size_t rowoff =
                    (size_t)(i0 + j * 16 + r) * H_DIM + k0 + half * 8;
                bfr[2 * j]     = ld_frag(GW + rowoff);
                bfr[2 * j + 1] = ld_frag(UW + rowoff);
            }
            bf16x16 afrag = ld_frag(&x_lds[r * XS + k0 + half * 8]);
            if (k0 + 32 < H_DIM) {
                __builtin_prefetch(GW + (size_t)(i0 + r) * H_DIM + k0 + 32, 0, 3);
                __builtin_prefetch(UW + (size_t)(i0 + r) * H_DIM + k0 + 32, 0, 3);
            }
#pragma unroll
            for (int j = 0; j < 4; ++j) {
                gacc[j] = wmma_bf16(afrag, bfr[2 * j],     gacc[j]);
                uacc[j] = wmma_bf16(afrag, bfr[2 * j + 1], uacc[j]);
            }
        }
        // h = silu(g)*u -> LDS [token m][i]
#pragma unroll
        for (int j = 0; j < 4; ++j) {
#pragma unroll
            for (int v = 0; v < 8; ++v) {
                float g = gacc[j][v];
                float hval = g * (1.f / (1.f + __expf(-g))) * uacc[j][v];
                int m = v + 8 * half;
                h_lds[m * HS + i0 + j * 16 + r] = f2bf_bits(hval);
            }
        }
        __syncthreads();

        // ---------- Phase B: out N-range [wave*128, +128), K over I ----------
        const int n0 = wave * 128;
        f32x8 c[8];
#pragma unroll
        for (int j = 0; j < 8; ++j) c[j] = (f32x8){};

        for (int k0 = 0; k0 < I_MOE; k0 += 32) {
            bf16x16 bfr[8];
#pragma unroll
            for (int j = 0; j < 8; ++j) {
                const unsigned short* bp =
                    DW + (size_t)(n0 + j * 16 + r) * drs + koff + k0 + half * 8;
                bfr[j] = ld_frag(bp);
            }
            bf16x16 afrag = ld_frag(&h_lds[r * HS + k0 + half * 8]);
#pragma unroll
            for (int j = 0; j < 8; ++j)
                c[j] = wmma_bf16(afrag, bfr[j], c[j]);
        }
#pragma unroll
        for (int j = 0; j < 8; ++j) {
#pragma unroll
            for (int v = 0; v < 8; ++v) {
                int m = v + 8 * half;
                float w = (ex < 16) ? comb_lds[m * N_EXP + ex] : 1.0f;
                oacc[j][v] += w * c[j][v];
            }
        }
        __syncthreads();   // h_lds reused next expert
    }

#pragma unroll
    for (int j = 0; j < 8; ++j) {
#pragma unroll
        for (int v = 0; v < 8; ++v) {
            int m = v + 8 * half;
            out[(size_t)(t0 + m) * H_DIM + wave * 128 + j * 16 + r] = oacc[j][v];
        }
    }
}

// =====================================================================
extern "C" void kernel_launch(void* const* d_in, const int* in_sizes, int n_in,
                              void* d_out, int out_size, void* d_ws, size_t ws_size,
                              hipStream_t stream) {
    const float* x     = (const float*)d_in[0];
    const float* rw    = (const float*)d_in[1];
    const float* ebias = (const float*)d_in[2];
    const float* gw    = (const float*)d_in[3];
    const float* uw    = (const float*)d_in[4];
    const float* dw    = (const float*)d_in[5];
    const float* shg   = (const float*)d_in[6];
    const float* shu   = (const float*)d_in[7];
    const float* shd   = (const float*)d_in[8];
    float* out = (float*)d_out;

    unsigned short* bfp  = (unsigned short*)d_ws;
    float*          comb = (float*)((char*)d_ws + COMB_OFF_BYTES);

    // one-time f32 -> bf16 staging of weights + activations (memory bound)
    auto cvt = [&](const float* s, unsigned long long off, unsigned long long n) {
        int n8 = (int)(n / 8);
        int grid = (n8 + 255) / 256; if (grid > 4096) grid = 4096;
        cvt_bf16_kernel<<<grid, 256, 0, stream>>>(s, bfp + off, n8);
    };
    cvt(gw,  OFF_GATE, OFF_UP   - OFF_GATE);
    cvt(uw,  OFF_UP,   OFF_DOWN - OFF_UP);
    cvt(dw,  OFF_DOWN, OFF_SHG  - OFF_DOWN);
    cvt(shg, OFF_SHG,  OFF_SHU  - OFF_SHG);
    cvt(shu, OFF_SHU,  OFF_SHD  - OFF_SHU);
    cvt(shd, OFF_SHD,  OFF_X    - OFF_SHD);
    cvt(x,   OFF_X,    TOTAL_BF - OFF_X);

    moe_router_kernel<<<T_TOK / 8, 256, 0, stream>>>(x, rw, ebias, comb);

    moe_ffn_kernel<<<T_TOK / 16, 256, 0, stream>>>(
        bfp + OFF_X, comb,
        bfp + OFF_GATE, bfp + OFF_UP, bfp + OFF_DOWN,
        bfp + OFF_SHG, bfp + OFF_SHU, bfp + OFF_SHD, out);
}
